// DifferentialMultiQueryAttention_27066883899739
// MI455X (gfx1250) — compile-verified, matrix operand order checked by reading, not run on hardware
//
#include <hip/hip_runtime.h>
#include <hip/hip_bf16.h>

// ---------------------------------------------------------------------------
// Problem constants (from the reference)
// ---------------------------------------------------------------------------
constexpr int kSeq    = 2048;
constexpr int kHidden = 2048;
constexpr int kHeads  = 16;
constexpr int kHDim   = 128;
#define LAMBDA 0.5f

// ---------------------------------------------------------------------------
// CDNA5 feature probes (compile-safe: manual fallbacks if builtins absent)
// ---------------------------------------------------------------------------
#if defined(__HIP_DEVICE_COMPILE__) && __has_builtin(__builtin_amdgcn_tensor_load_to_lds)
#define HAVE_TDM 1
#else
#define HAVE_TDM 0
#endif

#if defined(__HIP_DEVICE_COMPILE__) && __has_builtin(__builtin_amdgcn_global_load_async_to_lds_b128)
#define HAVE_ASYNC_LDS 1
#else
#define HAVE_ASYNC_LDS 0
#endif

// ---------------------------------------------------------------------------
// CDNA5 WMMA / descriptor types
// ---------------------------------------------------------------------------
typedef __bf16 bf16_t;
typedef __attribute__((ext_vector_type(16))) __bf16 v16bf;
typedef __attribute__((ext_vector_type(8)))  float  v8f;
typedef unsigned int u32x4 __attribute__((ext_vector_type(4)));
typedef int          i32x4 __attribute__((ext_vector_type(4)));
typedef int          i32x8 __attribute__((ext_vector_type(8)));

// Address-space qualified vector pointers for the async global->LDS builtins.
typedef __attribute__((address_space(1))) i32x4 gAS_i32x4;
typedef __attribute__((address_space(3))) i32x4 lAS_i32x4;

union FragBF {
    v16bf v;
    uint4 q[2];
    unsigned short s[16];
};

__device__ __forceinline__ unsigned short f2bf_u16(float f) {
    union { float f; unsigned u; } a; a.f = f;
    unsigned u = a.u;
    unsigned r = u + 0x7FFFu + ((u >> 16) & 1u);   // round-to-nearest-even
    return (unsigned short)(r >> 16);
}

// LDS byte offset of a generic pointer into __shared__ space.
__device__ __forceinline__ unsigned lds_offset(const void* p) {
#if defined(__HIP_DEVICE_COMPILE__)
    return (unsigned)(unsigned long long)(__attribute__((address_space(3))) const void*)p;
#else
    return 0;
#endif
}

__device__ __forceinline__ void wait_async_zero() {
#if HAVE_ASYNC_LDS
#if __has_builtin(__builtin_amdgcn_s_wait_asynccnt)
    __builtin_amdgcn_s_wait_asynccnt(0);
#else
    asm volatile("s_wait_asynccnt 0" ::: "memory");
#endif
#endif
}

// Load one 16x32 bf16 WMMA operand fragment from a row-major tile
// (matches the documented 16-bit A-matrix lane/VGPR layout; the dense B
// matrix mirrors it, so transposed-stored tiles reuse this loader).
__device__ __forceinline__ v16bf load_frag_rm(const unsigned short* p, int stride) {
    const int lane = threadIdx.x & 31;
    const int lr = lane & 15;
    const int hi = lane >> 4;
    FragBF f;
    const unsigned short* row = p + lr * stride;
    f.q[0] = *reinterpret_cast<const uint4*>(row + hi * 8);
    f.q[1] = *reinterpret_cast<const uint4*>(row + 16 + hi * 8);
    return f.v;
}

__device__ __forceinline__ v8f wmma_bf16(v16bf a, v16bf b, v8f c) {
    return __builtin_amdgcn_wmma_f32_16x16x32_bf16(false, a, false, b, (short)0, c,
                                                   false, false);
}

#if HAVE_TDM
// Issue one 2D Tensor-Data-Mover load: (tile_d0 x tile_d1) bf16 elements from a
// row-major global tensor into LDS, with TDM LDS padding producing 40-element
// rows (pad 4 DWORDs after every 16 DWORDs when tile_d0 == 32).
__device__ __forceinline__ void tdm_load_2d_to_lds(
    const void* gptr, unsigned ldsOff, int tensor_d0, int tensor_d1,
    int tile_d0, int tile_d1, int stride0, unsigned padbits)
{
    unsigned long long ga = (unsigned long long)gptr;
    u32x4 g0;
    g0.x = 1u;                                                 // count=1 (user D#)
    g0.y = ldsOff;                                             // lds_addr
    g0.z = (unsigned)ga;                                       // global_addr lo
    g0.w = (unsigned)((ga >> 32) & 0x01FFFFFFull) | 0x80000000u; // addr hi | type=2
    i32x8 g1;
    g1[0] = (int)((1u << 16) | padbits);                       // data_size=2B + pad cfg
    g1[1] = (int)(((unsigned)tensor_d0 & 0xFFFFu) << 16);      // tensor_dim0 lo16
    g1[2] = (int)(((unsigned)tensor_d0 >> 16) |
                  (((unsigned)tensor_d1 & 0xFFFFu) << 16));    // td0 hi | td1 lo
    g1[3] = (int)(((unsigned)tensor_d1 >> 16) |
                  ((unsigned)tile_d0 << 16));                  // td1 hi | tile_dim0
    g1[4] = (int)(unsigned)tile_d1;                            // tile_dim1 (dim2=0)
    g1[5] = stride0;                                           // tensor_dim0_stride
    g1[6] = 0;
    g1[7] = 0;
    i32x4 gz  = {0, 0, 0, 0};
    i32x8 gz8 = {0, 0, 0, 0, 0, 0, 0, 0};
    __builtin_amdgcn_tensor_load_to_lds(g0, g1, gz, gz, gz8, 0);
}
#endif

// ---------------------------------------------------------------------------
// fp32 -> bf16 conversion
// ---------------------------------------------------------------------------
__global__ void f32_to_bf16_kernel(const float* __restrict__ in,
                                   unsigned short* __restrict__ out, int n) {
    int i = blockIdx.x * blockDim.x + threadIdx.x;
    if (i < n) out[i] = f2bf_u16(in[i]);
}

// ---------------------------------------------------------------------------
// Tiled bf16 GEMM: C(MxN fp32) = A(MxK bf16, row-major) @ B(KxN bf16, row-major)
// Block = 128 threads (4 waves); block tile 128x64; each wave owns 32 rows.
// A tile staged by the TDM when available; B tile staged transposed manually.
// ---------------------------------------------------------------------------
__global__ __launch_bounds__(128) void gemm_bf16_kernel(
    const unsigned short* __restrict__ A, const unsigned short* __restrict__ B,
    float* __restrict__ C, int M, int N, int K)
{
    __shared__ __align__(16) unsigned short As[128 * 40];  // 128 rows x 32 K (pad->40)
    __shared__ __align__(16) unsigned short BsT[64 * 40];  // 64 cols x 32 K (transposed)

    const int tid  = threadIdx.x;
    const int wv   = tid >> 5;
    const int lane = tid & 31;
    const int lr   = lane & 15;
    const int hi   = lane >> 4;
    const int m0   = blockIdx.y * 128;
    const int n0   = blockIdx.x * 64;

    v8f acc[2][4];
    #pragma unroll
    for (int half = 0; half < 2; ++half)
        #pragma unroll
        for (int nt = 0; nt < 4; ++nt) acc[half][nt] = (v8f){};

#if HAVE_TDM
    const unsigned asOff = lds_offset(&As[0]);
    // pad_enable | pad_interval=16 DWORDs (enc 3) | pad_amount=4 DWORDs (enc 3)
    const unsigned padbits = (1u << 20) | (3u << 22) | (3u << 25);
#endif

    const int kchunks = K >> 5;
    for (int kc = 0; kc < kchunks; ++kc) {
        const int k0 = kc * 32;

#if HAVE_TDM
        if (wv == 0) {
            tdm_load_2d_to_lds(A + (size_t)m0 * K + k0, asOff,
                               K, M, 32, 128, K, padbits);
            __builtin_amdgcn_s_wait_tensorcnt(0);
        }
#else
        {   // A tile 128x32: 512 x 16B slots, 4 per thread, fully unrolled
            const int r = tid >> 2;          // 0..31
            const int c = (tid & 3) * 8;
            #pragma unroll
            for (int i = 0; i < 4; ++i) {
                uint4 a = *reinterpret_cast<const uint4*>(
                    A + (size_t)(m0 + r + 32 * i) * K + k0 + c);
                *reinterpret_cast<uint4*>(&As[(r + 32 * i) * 40 + c]) = a;
            }
        }
#endif
        {   // B tile 32x64 staged transposed -> BsT[col][k]
            const int rb = tid >> 3;         // 0..15
            const int cb = (tid & 7) * 8;
            union { uint4 q; unsigned short h[8]; } d0, d1;
            d0.q = *reinterpret_cast<const uint4*>(
                B + (size_t)(k0 + rb) * N + n0 + cb);
            d1.q = *reinterpret_cast<const uint4*>(
                B + (size_t)(k0 + rb + 16) * N + n0 + cb);
            #pragma unroll
            for (int e = 0; e < 8; ++e) {
                BsT[(cb + e) * 40 + rb]      = d0.h[e];
                BsT[(cb + e) * 40 + rb + 16] = d1.h[e];
            }
        }
        if (kc + 1 < kchunks) {  // global_prefetch_b8 of the next K chunk
            __builtin_prefetch(A + (size_t)(m0 + tid) * K + k0 + 32, 0, 1);
            __builtin_prefetch(B + (size_t)(k0 + 32 + (tid >> 2)) * N + n0, 0, 1);
        }
        __syncthreads();

        v16bf af0 = load_frag_rm(&As[(wv * 32) * 40], 40);
        v16bf af1 = load_frag_rm(&As[(wv * 32 + 16) * 40], 40);
        #pragma unroll
        for (int nt = 0; nt < 4; ++nt) {
            v16bf bfg = load_frag_rm(&BsT[(nt * 16) * 40], 40);
            acc[0][nt] = wmma_bf16(af0, bfg, acc[0][nt]);
            acc[1][nt] = wmma_bf16(af1, bfg, acc[1][nt]);
        }
        __syncthreads();
    }

    // D layout: lane -> column (n), VGPR j -> row j (+8 for high half-lanes)
    #pragma unroll
    for (int half = 0; half < 2; ++half)
        #pragma unroll
        for (int nt = 0; nt < 4; ++nt)
            #pragma unroll
            for (int j = 0; j < 8; ++j) {
                int row = m0 + wv * 32 + half * 16 + j + hi * 8;
                int col = n0 + nt * 16 + lr;
                C[(size_t)row * N + col] = acc[half][nt][j];
            }
}

// ---------------------------------------------------------------------------
// RoPE for q1/q2 (fp32 (S,H*D) -> bf16 (H,S,D)) and K/V (fp32 (S,D) -> bf16)
// ---------------------------------------------------------------------------
__global__ void rope_q_kernel(const float* __restrict__ q1f,
                              const float* __restrict__ q2f,
                              const float* __restrict__ fcos,
                              const float* __restrict__ fsin,
                              unsigned short* __restrict__ q1b,
                              unsigned short* __restrict__ q2b)
{
    const int s = blockIdx.x, h = blockIdx.y, d = threadIdx.x;
    const size_t in  = (size_t)s * (kHeads * kHDim) + h * kHDim + d;
    const size_t out = ((size_t)h * kSeq + s) * kHDim + d;
    const float c  = fcos[s * kHDim + d];
    const float sn = fsin[s * kHDim + d];
    float a1 = q1f[in];
    float r1 = (d < 64) ? -q1f[in + 64] : q1f[in - 64];
    float a2 = q2f[in];
    float r2 = (d < 64) ? -q2f[in + 64] : q2f[in - 64];
    q1b[out] = f2bf_u16(a1 * c + r1 * sn);
    q2b[out] = f2bf_u16(a2 * c + r2 * sn);
}

__global__ void rope_kv_kernel(const float* __restrict__ kf,
                               const float* __restrict__ vf,
                               const float* __restrict__ fcos,
                               const float* __restrict__ fsin,
                               unsigned short* __restrict__ kbo,
                               unsigned short* __restrict__ vbo)
{
    const int s = blockIdx.x, d = threadIdx.x;
    const size_t i = (size_t)s * kHDim + d;
    const float c  = fcos[i];
    const float sn = fsin[i];
    float a = kf[i];
    float r = (d < 64) ? -kf[i + 64] : kf[i - 64];
    kbo[i] = f2bf_u16(a * c + r * sn);
    vbo[i] = f2bf_u16(vf[i]);
}

// ---------------------------------------------------------------------------
// Fused causal differential attention.
// Grid (S/64, HEADS), block 128 (4 waves). Wave w owns q rows [qt, qt+16).
// Pass A: online softmax stats (m,Z) for s1 and s2 via WMMA score tiles.
// Pass B: p = relu((e1/Z1 - lam*e2/Z2)/(1-lam+1e-8)); out += P @ V via WMMA.
// K tiles staged with async global->LDS loads (ASYNCcnt) when available.
// ---------------------------------------------------------------------------
__global__ __launch_bounds__(128) void diff_attn_kernel(
    const unsigned short* __restrict__ q1b,  // (H,S,D) bf16
    const unsigned short* __restrict__ q2b,  // (H,S,D) bf16
    const unsigned short* __restrict__ kb,   // (S,D)   bf16
    const unsigned short* __restrict__ vb,   // (S,D)   bf16
    unsigned short* __restrict__ attn)       // (S,H*D) bf16
{
    __shared__ __align__(16) unsigned short kbuf[32 * 136];   // 32 keys x 128 dims
    __shared__ __align__(16) unsigned short vT[128 * 40];     // 128 dims x 32 keys
    __shared__ __align__(16) unsigned short pbuf[4][16 * 40]; // per-wave P tile

    const int tid  = threadIdx.x;
    const int wv   = tid >> 5;
    const int lane = tid & 31;
    const int lr   = lane & 15;
    const int hi   = lane >> 4;
    const int h    = blockIdx.y;
    const int qt   = blockIdx.x * 64 + wv * 16;

    // q fragments live in registers for the whole kernel (16 rows x 128 dims x2)
    v16bf aq1[4], aq2[4];
    {
        const unsigned short* q1p = q1b + ((size_t)h * kSeq + qt) * kHDim;
        const unsigned short* q2p = q2b + ((size_t)h * kSeq + qt) * kHDim;
        #pragma unroll
        for (int kc = 0; kc < 4; ++kc) {
            aq1[kc] = load_frag_rm(q1p + kc * 32, kHDim);
            aq2[kc] = load_frag_rm(q2p + kc * 32, kHDim);
        }
    }

    const float scale = 0.08838834764831845f;  // 1/sqrt(128)
    float m1[8], z1[8], m2[8], z2[8];
    #pragma unroll
    for (int j = 0; j < 8; ++j) { m1[j] = -1e30f; z1[j] = 0.f; m2[j] = -1e30f; z2[j] = 0.f; }

    const int nchunks  = blockIdx.x * 2 + 2;       // block-uniform causal bound
    const int myChunks = (qt + 15) / 32 + 1;       // wave-uniform causal bound

    const int cb = (tid & 15) * 8;                 // staging slot: column chunk
    const int rbs = tid >> 4;                      // staging slot: base row (0..7)

    // ---------------- Pass A: softmax statistics ----------------
    for (int c = 0; c < nchunks; ++c) {
        const int kb0 = c * 32;
        #pragma unroll
        for (int i = 0; i < 4; ++i) {              // stage 32x128 K tile
            const int r = rbs + 8 * i;
            const unsigned short* src = kb + (size_t)(kb0 + r) * kHDim + cb;
#if HAVE_ASYNC_LDS
            __builtin_amdgcn_global_load_async_to_lds_b128(
                (gAS_i32x4*)src, (lAS_i32x4*)&kbuf[r * 136 + cb], 0, 0);
#else
            *reinterpret_cast<uint4*>(&kbuf[r * 136 + cb]) =
                *reinterpret_cast<const uint4*>(src);
#endif
        }
        wait_async_zero();
        __syncthreads();

        if (c < myChunks) {
            #pragma unroll
            for (int t = 0; t < 2; ++t) {
                v8f c1 = (v8f){}, c2 = (v8f){};
                #pragma unroll
                for (int kc = 0; kc < 4; ++kc) {
                    v16bf bk = load_frag_rm(&kbuf[(t * 16) * 136 + kc * 32], 136);
                    c1 = wmma_bf16(aq1[kc], bk, c1);
                    c2 = wmma_bf16(aq2[kc], bk, c2);
                }
                const int key = kb0 + t * 16 + lr;
                #pragma unroll
                for (int j = 0; j < 8; ++j) {
                    const int qrow = qt + j + hi * 8;
                    const bool ok = (key <= qrow);
                    float s1 = ok ? c1[j] * scale : -1e10f;
                    float s2 = ok ? c2[j] * scale : -1e10f;
                    float t1 = s1, t2 = s2;
                    #pragma unroll
                    for (int o = 1; o < 16; o <<= 1) {
                        t1 = fmaxf(t1, __shfl_xor(t1, o, 32));
                        t2 = fmaxf(t2, __shfl_xor(t2, o, 32));
                    }
                    float mn1 = fmaxf(m1[j], t1);
                    float mn2 = fmaxf(m2[j], t2);
                    float e1 = __expf(s1 - mn1);
                    float e2 = __expf(s2 - mn2);
                    #pragma unroll
                    for (int o = 1; o < 16; o <<= 1) {
                        e1 += __shfl_xor(e1, o, 32);
                        e2 += __shfl_xor(e2, o, 32);
                    }
                    z1[j] = z1[j] * __expf(m1[j] - mn1) + e1;
                    z2[j] = z2[j] * __expf(m2[j] - mn2) + e2;
                    m1[j] = mn1;
                    m2[j] = mn2;
                }
            }
        }
        __syncthreads();
    }

    // Row-sum of (a1 - lam*a2) is analytically (1 - lam).
    const float rdenom = 1.0f / (1.0f - LAMBDA + 1e-8f);
    float rz1[8], rz2[8];
    #pragma unroll
    for (int j = 0; j < 8; ++j) { rz1[j] = 1.0f / z1[j]; rz2[j] = LAMBDA / z2[j]; }

    v8f acc[8];
    #pragma unroll
    for (int nt = 0; nt < 8; ++nt) acc[nt] = (v8f){};

    // ---------------- Pass B: P = relu(diff-softmax); out += P @ V ----------
    for (int c = 0; c < nchunks; ++c) {
        const int kb0 = c * 32;
        #pragma unroll
        for (int i = 0; i < 4; ++i) {              // stage K (async) + V transposed
            const int r = rbs + 8 * i;
            const unsigned short* ksrc = kb + (size_t)(kb0 + r) * kHDim + cb;
#if HAVE_ASYNC_LDS
            __builtin_amdgcn_global_load_async_to_lds_b128(
                (gAS_i32x4*)ksrc, (lAS_i32x4*)&kbuf[r * 136 + cb], 0, 0);
#else
            *reinterpret_cast<uint4*>(&kbuf[r * 136 + cb]) =
                *reinterpret_cast<const uint4*>(ksrc);
#endif
            union { uint4 q; unsigned short hh[8]; } u;
            u.q = *reinterpret_cast<const uint4*>(
                vb + (size_t)(kb0 + r) * kHDim + cb);
            #pragma unroll
            for (int e = 0; e < 8; ++e) vT[(cb + e) * 40 + r] = u.hh[e];
        }
        wait_async_zero();
        __syncthreads();

        if (c < myChunks) {
            #pragma unroll
            for (int t = 0; t < 2; ++t) {
                v8f c1 = (v8f){}, c2 = (v8f){};
                #pragma unroll
                for (int kc = 0; kc < 4; ++kc) {
                    v16bf bk = load_frag_rm(&kbuf[(t * 16) * 136 + kc * 32], 136);
                    c1 = wmma_bf16(aq1[kc], bk, c1);
                    c2 = wmma_bf16(aq2[kc], bk, c2);
                }
                const int key = kb0 + t * 16 + lr;
                #pragma unroll
                for (int j = 0; j < 8; ++j) {
                    const int qrow = qt + j + hi * 8;
                    float p = 0.0f;
                    if (key <= qrow) {
                        float e1 = __expf(c1[j] * scale - m1[j]) * rz1[j];
                        float e2 = __expf(c2[j] * scale - m2[j]) * rz2[j];
                        p = fmaxf((e1 - e2) * rdenom, 0.0f);
                    }
                    pbuf[wv][(j + hi * 8) * 40 + t * 16 + lr] = f2bf_u16(p);
                }
            }
            // P tile (16x32) re-read in A layout; V fragments from transposed tile
            v16bf ap = load_frag_rm(&pbuf[wv][0], 40);
            #pragma unroll
            for (int nt = 0; nt < 8; ++nt) {
                v16bf bv = load_frag_rm(&vT[(nt * 16) * 40], 40);
                acc[nt] = wmma_bf16(ap, bv, acc[nt]);
            }
        }
        __syncthreads();
    }

    #pragma unroll
    for (int nt = 0; nt < 8; ++nt)
        #pragma unroll
        for (int j = 0; j < 8; ++j) {
            int row = qt + j + hi * 8;
            int col = h * kHDim + nt * 16 + lr;
            attn[(size_t)row * (kHeads * kHDim) + col] = f2bf_u16(acc[nt][j]);
        }
}

// ---------------------------------------------------------------------------
// Host-side orchestration
// ---------------------------------------------------------------------------
extern "C" void kernel_launch(void* const* d_in, const int* in_sizes, int n_in,
                              void* d_out, int out_size, void* d_ws, size_t ws_size,
                              hipStream_t stream) {
    const float* x   = (const float*)d_in[0];
    const float* fc  = (const float*)d_in[1];
    const float* fs  = (const float*)d_in[2];
    /* d_in[3] = causal mask (tril) -- implied by kernel, unused */
    const float* q1w = (const float*)d_in[4];
    const float* q2w = (const float*)d_in[5];
    const float* kw  = (const float*)d_in[6];
    const float* vw  = (const float*)d_in[7];
    const float* ow  = (const float*)d_in[8];
    float* out = (float*)d_out;

    const int nXW = kSeq * kHidden;          // 2048*2048
    const int nKV = kHidden * kHDim;         // 2048*128

    char* ws = (char*)d_ws;
    auto carve = [&](size_t bytes) {
        char* p = ws;
        ws += (bytes + 255) & ~(size_t)255;
        return p;
    };
    unsigned short* xb   = (unsigned short*)carve((size_t)nXW * 2);
    unsigned short* wq1b = (unsigned short*)carve((size_t)nXW * 2);
    unsigned short* wq2b = (unsigned short*)carve((size_t)nXW * 2);
    unsigned short* wob  = (unsigned short*)carve((size_t)nXW * 2);
    unsigned short* wkb  = (unsigned short*)carve((size_t)nKV * 2);
    unsigned short* wvb  = (unsigned short*)carve((size_t)nKV * 2);
    float*          q1f  = (float*)carve((size_t)nXW * 4);
    float*          q2f  = (float*)carve((size_t)nXW * 4);
    float*          kf   = (float*)carve((size_t)kSeq * kHDim * 4);
    float*          vf   = (float*)carve((size_t)kSeq * kHDim * 4);
    unsigned short* q1r  = (unsigned short*)carve((size_t)nXW * 2);
    unsigned short* q2r  = (unsigned short*)carve((size_t)nXW * 2);
    unsigned short* kr   = (unsigned short*)carve((size_t)kSeq * kHDim * 2);
    unsigned short* vr   = (unsigned short*)carve((size_t)kSeq * kHDim * 2);
    unsigned short* attn = (unsigned short*)carve((size_t)nXW * 2);

    // 1) fp32 -> bf16 for activations + weights
    f32_to_bf16_kernel<<<(nXW + 255) / 256, 256, 0, stream>>>(x,   xb,   nXW);
    f32_to_bf16_kernel<<<(nXW + 255) / 256, 256, 0, stream>>>(q1w, wq1b, nXW);
    f32_to_bf16_kernel<<<(nXW + 255) / 256, 256, 0, stream>>>(q2w, wq2b, nXW);
    f32_to_bf16_kernel<<<(nXW + 255) / 256, 256, 0, stream>>>(ow,  wob,  nXW);
    f32_to_bf16_kernel<<<(nKV + 255) / 256, 256, 0, stream>>>(kw,  wkb,  nKV);
    f32_to_bf16_kernel<<<(nKV + 255) / 256, 256, 0, stream>>>(vw,  wvb,  nKV);

    // 2) projections (bf16 WMMA, fp32 accumulate); block tile 128x64
    gemm_bf16_kernel<<<dim3(kHidden / 64, kSeq / 128), 128, 0, stream>>>(
        xb, wq1b, q1f, kSeq, kHidden, kHidden);
    gemm_bf16_kernel<<<dim3(kHidden / 64, kSeq / 128), 128, 0, stream>>>(
        xb, wq2b, q2f, kSeq, kHidden, kHidden);
    gemm_bf16_kernel<<<dim3(kHDim / 64, kSeq / 128), 128, 0, stream>>>(
        xb, wkb, kf, kSeq, kHDim, kHidden);
    gemm_bf16_kernel<<<dim3(kHDim / 64, kSeq / 128), 128, 0, stream>>>(
        xb, wvb, vf, kSeq, kHDim, kHidden);

    // 3) RoPE (fp32 math) -> bf16 attention operands
    rope_q_kernel<<<dim3(kSeq, kHeads), kHDim, 0, stream>>>(q1f, q2f, fc, fs, q1r, q2r);
    rope_kv_kernel<<<dim3(kSeq), kHDim, 0, stream>>>(kf, vf, fc, fs, kr, vr);

    // 4) fused causal differential attention
    diff_attn_kernel<<<dim3(kSeq / 64, kHeads), 128, 0, stream>>>(
        q1r, q2r, kr, vr, attn);

    // 5) output projection -> fp32 d_out
    gemm_bf16_kernel<<<dim3(kHidden / 64, kSeq / 128), 128, 0, stream>>>(
        attn, wob, out, kSeq, kHidden, kHidden);
}
